// SingleStreamBlock_85126251806953
// MI455X (gfx1250) — compile-verified, matrix operand order checked by reading, not run on hardware
//
#include <hip/hip_runtime.h>
#include <hip/hip_bf16.h>
#include <math.h>

// ---- types for WMMA (CDNA5 / gfx1250, wave32) ----
typedef __bf16 v16bf __attribute__((ext_vector_type(16)));
typedef __bf16 v8bf  __attribute__((ext_vector_type(8)));
typedef float  v8f   __attribute__((ext_vector_type(8)));

union FragBF { v16bf v; v8bf h[2]; };

#define WMMA_BF16(a, b, c) \
  __builtin_amdgcn_wmma_f32_16x16x32_bf16(false, (a), false, (b), (short)0, (c), false, false)

// ---- gfx1250 async direct-to-LDS path (guarded: falls back to sync copy) ----
#if defined(__AMDGCN__) && __has_builtin(__builtin_amdgcn_global_load_async_to_lds_b128)
#define HAS_ASYNC 1
#else
#define HAS_ASYNC 0
#endif

#if HAS_ASYNC
typedef int b128v __attribute__((vector_size(16)));
typedef __attribute__((address_space(1))) b128v AS1_b128;  // global
typedef __attribute__((address_space(3))) b128v AS3_b128;  // LDS
#endif

// copy 8 bf16 (16 bytes) global -> LDS
__device__ __forceinline__ void cp16(const __bf16* g, __bf16* l) {
#if HAS_ASYNC
  __builtin_amdgcn_global_load_async_to_lds_b128(
      (AS1_b128*)(unsigned long long)g,
      (AS3_b128*)(unsigned int)(unsigned long long)l, 0, 0);
#else
  *(v8bf*)l = *(const v8bf*)g;
#endif
}

__device__ __forceinline__ void cp_wait() {
#if HAS_ASYNC
#if __has_builtin(__builtin_amdgcn_s_wait_asynccnt)
  __builtin_amdgcn_s_wait_asynccnt(0);
#else
  asm volatile("s_wait_asynccnt 0x0" ::: "memory");
#endif
#endif
}

// Problem constants
// B=1, L=2048, D=3072, H=24, Dh=128, MLP=12288, 3D+MLP=21504, D+MLP=15360
#define L_SEQ 2048
#define D_MODEL 3072
#define N_HEAD 24
#define D_HEAD 128
#define NC1 21504
#define KC2 15360
#define QKV3 9216

// ---------------------------------------------------------------
// fp32 -> bf16 conversion (grid-stride)
// ---------------------------------------------------------------
__global__ __launch_bounds__(256) void k_conv(const float* __restrict__ in,
                                              __bf16* __restrict__ out, size_t n) {
  for (size_t i = (size_t)blockIdx.x * 256 + threadIdx.x; i < n;
       i += (size_t)gridDim.x * 256)
    out[i] = (__bf16)in[i];
}

// ---------------------------------------------------------------
// mod = silu(vec) @ mod_w + mod_b   (1x3072 @ 3072x9216)
// ---------------------------------------------------------------
__global__ __launch_bounds__(256) void k_mod(const float* __restrict__ vec,
                                             const float* __restrict__ mod_w,
                                             const float* __restrict__ mod_b,
                                             float* __restrict__ modv) {
  __shared__ float sv[D_MODEL];
  int tid = threadIdx.x;
  for (int i = tid; i < D_MODEL; i += 256) {
    float x = vec[i];
    sv[i] = x / (1.f + __expf(-x));
  }
  __syncthreads();
  int n = blockIdx.x * 256 + tid;
  float acc = mod_b[n];
  for (int k = 0; k < D_MODEL; ++k)
    acc += sv[k] * mod_w[(size_t)k * QKV3 + n];
  modv[n] = acc;
}

// ---------------------------------------------------------------
// LayerNorm + modulation -> bf16 activations
// ---------------------------------------------------------------
__global__ __launch_bounds__(256) void k_ln(const float* __restrict__ img,
                                            const float* __restrict__ modv,
                                            __bf16* __restrict__ xb) {
  int row = blockIdx.x, tid = threadIdx.x;
  int wave = tid >> 5, lane = tid & 31;
  const float* p = img + (size_t)row * D_MODEL;
  float v[12], s = 0.f, s2 = 0.f;
#pragma unroll
  for (int i = 0; i < 12; ++i) {
    float x = p[tid + i * 256];
    v[i] = x; s += x; s2 += x * x;
  }
#pragma unroll
  for (int m = 1; m < 32; m <<= 1) {
    s  += __shfl_xor(s,  m, 32);
    s2 += __shfl_xor(s2, m, 32);
  }
  __shared__ float rs[8], rs2[8];
  __shared__ float mu_s, rv_s;
  if (lane == 0) { rs[wave] = s; rs2[wave] = s2; }
  __syncthreads();
  if (tid == 0) {
    float a = 0.f, b = 0.f;
    for (int i = 0; i < 8; ++i) { a += rs[i]; b += rs2[i]; }
    float mu = a / (float)D_MODEL;
    float var = b / (float)D_MODEL - mu * mu;
    mu_s = mu;
    rv_s = rsqrtf(var + 1e-6f);
  }
  __syncthreads();
  float mu = mu_s, rv = rv_s;
  __bf16* xo = xb + (size_t)row * D_MODEL;
#pragma unroll
  for (int i = 0; i < 12; ++i) {
    int d = tid + i * 256;
    float xn = (v[i] - mu) * rv;
    float y = (1.f + modv[D_MODEL + d]) * xn + modv[d];
    xo[d] = (__bf16)y;
  }
}

// ---------------------------------------------------------------
// bf16 WMMA GEMM: C(L x NDIM) = A(L x KDIM) @ Bw(KDIM x NDIM) + bias
// Block tile 128x128, 8 waves (2x4), each wave 64x32 = 4x2 WMMA tiles.
// A tile staged via async-to-LDS (ASYNCcnt) when available.
// MODE 1: lin1 epilogue (qkv fp32 / gelu->bf16 into cat)
// MODE 2: lin2 epilogue (img + gate*out -> fp32)
// ---------------------------------------------------------------
template <int KDIM, int NDIM, int MODE>
__global__ __launch_bounds__(256) void k_gemm(const __bf16* __restrict__ A,
                                              const __bf16* __restrict__ Bw,
                                              const float* __restrict__ bias,
                                              float* __restrict__ qkvf,
                                              __bf16* __restrict__ cat,
                                              const float* __restrict__ img,
                                              const float* __restrict__ modv,
                                              float* __restrict__ outp) {
  __shared__ __align__(16) __bf16 As[128][40];    // A tile, row-major (K pad)
  __shared__ __align__(16) __bf16 Bst[128][40];   // B tile transposed [col][k]
  int tid = threadIdx.x, wave = tid >> 5, lane = tid & 31;
  int wm = wave >> 2, wn = wave & 3;
  int lm = lane & 15, lg = lane >> 4;
  int m0 = blockIdx.y * 128, n0 = blockIdx.x * 128;
  v8f acc[4][2] = {};

  int ar = tid >> 1, ac = (tid & 1) * 16;     // A loader: row, col-half
  int bk = tid >> 3, bc = (tid & 7) * 16;     // B loader: k-row, col-base

  for (int k0 = 0; k0 < KDIM; k0 += 32) {
    // stage A (128x32) -- direct-to-LDS async when available
    const __bf16* ga = A + (size_t)(m0 + ar) * KDIM + k0 + ac;
    cp16(ga,     &As[ar][ac]);
    cp16(ga + 8, &As[ar][ac + 8]);
    // stage B transposed (32x128 -> [col][k]) via registers
    const v8bf* gb = (const v8bf*)(Bw + (size_t)(k0 + bk) * NDIM + n0 + bc);
    v8bf b0 = gb[0], b1 = gb[1];
#pragma unroll
    for (int j = 0; j < 8; ++j) {
      Bst[bc + j][bk]     = b0[j];
      Bst[bc + 8 + j][bk] = b1[j];
    }
    if (k0 + 32 < KDIM) {
      __builtin_prefetch(A + (size_t)(m0 + ar) * KDIM + k0 + 32 + ac, 0, 3);
      __builtin_prefetch(Bw + (size_t)(k0 + 32 + bk) * NDIM + n0 + bc, 0, 3);
    }
    cp_wait();
    __syncthreads();

    int koff = lg * 8;  // lane-group K pattern {0..7,16..23} / {8..15,24..31}
    FragBF afr[4], bfr[2];
#pragma unroll
    for (int mt = 0; mt < 4; ++mt) {
      int rA = wm * 64 + mt * 16 + lm;
      afr[mt].h[0] = *(const v8bf*)&As[rA][koff];
      afr[mt].h[1] = *(const v8bf*)&As[rA][koff + 16];
    }
#pragma unroll
    for (int nt = 0; nt < 2; ++nt) {
      int cB = wn * 32 + nt * 16 + lm;
      bfr[nt].h[0] = *(const v8bf*)&Bst[cB][koff];
      bfr[nt].h[1] = *(const v8bf*)&Bst[cB][koff + 16];
    }
#pragma unroll
    for (int mt = 0; mt < 4; ++mt)
#pragma unroll
      for (int nt = 0; nt < 2; ++nt)
        acc[mt][nt] = WMMA_BF16(afr[mt].v, bfr[nt].v, acc[mt][nt]);
    __syncthreads();
  }

  // epilogue: C layout -> lane l holds (row r+8*lg, col lm) in vgpr r
#pragma unroll
  for (int mt = 0; mt < 4; ++mt)
#pragma unroll
    for (int nt = 0; nt < 2; ++nt) {
      int col = n0 + wn * 32 + nt * 16 + lm;
      int rbase = m0 + wm * 64 + mt * 16 + 8 * lg;
      float bval = bias[col];
#pragma unroll
      for (int r = 0; r < 8; ++r) {
        int row = rbase + r;
        float v = acc[mt][nt][r] + bval;
        if constexpr (MODE == 1) {
          if (col < QKV3) {
            qkvf[(size_t)row * QKV3 + col] = v;
          } else {
            float g = 0.5f * v *
                      (1.f + tanhf(0.7978845608f * (v + 0.044715f * v * v * v)));
            cat[(size_t)row * KC2 + D_MODEL + (col - QKV3)] = (__bf16)g;
          }
        } else {
          float res = img[(size_t)row * D_MODEL + col] + modv[2 * D_MODEL + col] * v;
          outp[(size_t)row * D_MODEL + col] = res;
        }
      }
    }
}

// ---------------------------------------------------------------
// QK RMS-norm + RoPE -> head-major bf16 Q/K; V -> bf16
// one wave per (head, seq) row; lane handles 4 contiguous d (2 rope pairs)
// ---------------------------------------------------------------
__global__ __launch_bounds__(256) void k_qkrope(const float* __restrict__ qkvf,
                                                const float* __restrict__ pe,
                                                const float* __restrict__ qsc,
                                                const float* __restrict__ ksc,
                                                __bf16* __restrict__ qb,
                                                __bf16* __restrict__ kb,
                                                __bf16* __restrict__ vb) {
  int wave = threadIdx.x >> 5, lane = threadIdx.x & 31;
  int gid = blockIdx.x * 8 + wave;
  int h = gid >> 11;          // /2048
  int l = gid & 2047;
  const float* base = qkvf + (size_t)l * QKV3 + h * D_HEAD;
  int d0 = lane * 4;

#pragma unroll
  for (int which = 0; which < 2; ++which) {
    const float* src = base + which * D_MODEL;
    const float* sc = which ? ksc : qsc;
    __bf16* dst = (which ? kb : qb) + ((size_t)h * L_SEQ + l) * D_HEAD;
    float x0 = src[d0], x1 = src[d0 + 1], x2 = src[d0 + 2], x3 = src[d0 + 3];
    float ss = x0 * x0 + x1 * x1 + x2 * x2 + x3 * x3;
#pragma unroll
    for (int m = 1; m < 32; m <<= 1) ss += __shfl_xor(ss, m, 32);
    float r = rsqrtf(ss / (float)D_HEAD + 1e-6f);
    x0 *= r * sc[d0]; x1 *= r * sc[d0 + 1];
    x2 *= r * sc[d0 + 2]; x3 *= r * sc[d0 + 3];
    // pe[(l,p,j,i)] flat: (l*64+p)*4 + j*2 + i
    const float* pp = pe + ((size_t)l * 64 + (d0 >> 1)) * 4;
    float o0 = pp[0] * x0 + pp[1] * x1;
    float o1 = pp[2] * x0 + pp[3] * x1;
    float o2 = pp[4] * x2 + pp[5] * x3;
    float o3 = pp[6] * x2 + pp[7] * x3;
    dst[d0] = (__bf16)o0; dst[d0 + 1] = (__bf16)o1;
    dst[d0 + 2] = (__bf16)o2; dst[d0 + 3] = (__bf16)o3;
  }
  const float* vsrc = base + 2 * D_MODEL;
  __bf16* vdst = vb + ((size_t)h * L_SEQ + l) * D_HEAD;
  vdst[d0] = (__bf16)vsrc[d0];
  vdst[d0 + 1] = (__bf16)vsrc[d0 + 1];
  vdst[d0 + 2] = (__bf16)vsrc[d0 + 2];
  vdst[d0 + 3] = (__bf16)vsrc[d0 + 3];
}

// ---------------------------------------------------------------
// Flash attention: block = (head, 128 q rows), 8 waves x 16 q rows.
// 32-key tiles: S = Q@K^T (8 WMMA), online softmax, O += P@V (8 WMMA).
// K tile staged via async-to-LDS when available.
// ---------------------------------------------------------------
__global__ __launch_bounds__(256) void k_attn(const __bf16* __restrict__ qb,
                                              const __bf16* __restrict__ kb,
                                              const __bf16* __restrict__ vb,
                                              __bf16* __restrict__ cat) {
  __shared__ __align__(16) __bf16 Ks[32][136];     // K tile [key][d]
  __shared__ __align__(16) __bf16 Vst[128][40];    // V tile transposed [d][key]
  __shared__ __align__(16) __bf16 Ps[8][16][40];   // P re-layout per wave
  int tid = threadIdx.x, wave = tid >> 5, lane = tid & 31;
  int lm = lane & 15, lg = lane >> 4;
  int h = blockIdx.y, q0 = blockIdx.x * 128;
  int qw0 = q0 + wave * 16;
  const float sm = 0.08838834764831845f;  // 1/sqrt(128)

  FragBF qf[4];
  {
    size_t qrow = ((size_t)h * L_SEQ + qw0 + lm) * D_HEAD;
#pragma unroll
    for (int s = 0; s < 4; ++s) {
      size_t b = qrow + s * 32 + lg * 8;
      qf[s].h[0] = *(const v8bf*)(qb + b);
      qf[s].h[1] = *(const v8bf*)(qb + b + 16);
    }
  }

  v8f ov[8] = {};
  float mrow[8], srow[8];
#pragma unroll
  for (int r = 0; r < 8; ++r) { mrow[r] = -1e30f; srow[r] = 0.f; }

  int kj = tid >> 3, kc = (tid & 7) * 16;
  for (int kb0 = 0; kb0 < L_SEQ; kb0 += 32) {
    // stage K (async-to-LDS) and V (register transpose)
    {
      size_t grow = ((size_t)h * L_SEQ + kb0 + kj) * D_HEAD + kc;
      cp16(kb + grow,     &Ks[kj][kc]);
      cp16(kb + grow + 8, &Ks[kj][kc + 8]);
      const v8bf* gv = (const v8bf*)(vb + grow);
      v8bf v0v = gv[0], v1v = gv[1];
#pragma unroll
      for (int j = 0; j < 8; ++j) {
        Vst[kc + j][kj]     = v0v[j];
        Vst[kc + 8 + j][kj] = v1v[j];
      }
    }
    cp_wait();
    __syncthreads();

    // S = Q @ K^T  (contraction over d=128, 4 slices of 32)
    v8f s0 = {}, s1 = {};
#pragma unroll
    for (int s = 0; s < 4; ++s) {
      int doff = s * 32 + lg * 8;
      FragBF bk0, bk1;
      bk0.h[0] = *(const v8bf*)&Ks[lm][doff];
      bk0.h[1] = *(const v8bf*)&Ks[lm][doff + 16];
      bk1.h[0] = *(const v8bf*)&Ks[16 + lm][doff];
      bk1.h[1] = *(const v8bf*)&Ks[16 + lm][doff + 16];
      s0 = WMMA_BF16(qf[s].v, bk0.v, s0);
      s1 = WMMA_BF16(qf[s].v, bk1.v, s1);
    }

    // online softmax (row = r + 8*lg, spread over 16-lane group)
#pragma unroll
    for (int r = 0; r < 8; ++r) {
      float a0 = s0[r] * sm, a1 = s1[r] * sm;
      float mx = fmaxf(a0, a1);
#pragma unroll
      for (int m = 1; m < 16; m <<= 1) mx = fmaxf(mx, __shfl_xor(mx, m, 32));
      float nm = fmaxf(mrow[r], mx);
      float fac = __expf(mrow[r] - nm);
      float p0 = __expf(a0 - nm), p1 = __expf(a1 - nm);
      float rs = p0 + p1;
#pragma unroll
      for (int m = 1; m < 16; m <<= 1) rs += __shfl_xor(rs, m, 32);
      srow[r] = srow[r] * fac + rs;
      mrow[r] = nm;
#pragma unroll
      for (int t = 0; t < 8; ++t) ov[t][r] *= fac;
      int prow = r + 8 * lg;
      Ps[wave][prow][lm]      = (__bf16)p0;
      Ps[wave][prow][16 + lm] = (__bf16)p1;
    }
    asm volatile("s_wait_dscnt 0x0" ::: "memory");

    // O += P @ V  (contraction over 32 keys)
    FragBF ap;
    ap.h[0] = *(const v8bf*)&Ps[wave][lm][lg * 8];
    ap.h[1] = *(const v8bf*)&Ps[wave][lm][lg * 8 + 16];
#pragma unroll
    for (int t = 0; t < 8; ++t) {
      FragBF bv;
      int dcol = t * 16 + lm;
      bv.h[0] = *(const v8bf*)&Vst[dcol][lg * 8];
      bv.h[1] = *(const v8bf*)&Vst[dcol][lg * 8 + 16];
      ov[t] = WMMA_BF16(ap.v, bv.v, ov[t]);
    }
    __syncthreads();
  }

  // write attn output (bf16 into cat[:, :D])
#pragma unroll
  for (int t = 0; t < 8; ++t)
#pragma unroll
    for (int r = 0; r < 8; ++r) {
      int lqr = qw0 + r + 8 * lg;
      int d = t * 16 + lm;
      float val = ov[t][r] / srow[r];
      cat[(size_t)lqr * KC2 + h * D_HEAD + d] = (__bf16)val;
    }
}

// ---------------------------------------------------------------
extern "C" void kernel_launch(void* const* d_in, const int* in_sizes, int n_in,
                              void* d_out, int out_size, void* d_ws, size_t ws_size,
                              hipStream_t stream) {
  (void)in_sizes; (void)n_in; (void)out_size; (void)ws_size;
  const float* img    = (const float*)d_in[0];
  const float* vec    = (const float*)d_in[1];
  const float* pe     = (const float*)d_in[2];
  const float* mod_w  = (const float*)d_in[3];
  const float* mod_b  = (const float*)d_in[4];
  const float* lin1_w = (const float*)d_in[5];
  const float* lin1_b = (const float*)d_in[6];
  const float* lin2_w = (const float*)d_in[7];
  const float* lin2_b = (const float*)d_in[8];
  const float* q_sc   = (const float*)d_in[9];
  const float* k_sc   = (const float*)d_in[10];

  size_t o = 0;
  char* wsb = (char*)d_ws;
  auto nb = [&](size_t bytes) -> char* {
    char* p = wsb + o;
    o = (o + bytes + 255) & ~(size_t)255;
    return p;
  };
  __bf16* lin1wb = (__bf16*)nb((size_t)D_MODEL * NC1 * 2);
  __bf16* lin2wb = (__bf16*)nb((size_t)KC2 * D_MODEL * 2);
  __bf16* xb     = (__bf16*)nb((size_t)L_SEQ * D_MODEL * 2);
  float*  qkvf   = (float*) nb((size_t)L_SEQ * QKV3 * 4);
  __bf16* qb     = (__bf16*)nb((size_t)N_HEAD * L_SEQ * D_HEAD * 2);
  __bf16* kbv    = (__bf16*)nb((size_t)N_HEAD * L_SEQ * D_HEAD * 2);
  __bf16* vbv    = (__bf16*)nb((size_t)N_HEAD * L_SEQ * D_HEAD * 2);
  __bf16* catb   = (__bf16*)nb((size_t)L_SEQ * KC2 * 2);
  float*  modv   = (float*) nb((size_t)QKV3 * 4);

  k_conv<<<2048, 256, 0, stream>>>(lin1_w, lin1wb, (size_t)D_MODEL * NC1);
  k_conv<<<2048, 256, 0, stream>>>(lin2_w, lin2wb, (size_t)KC2 * D_MODEL);
  k_mod<<<QKV3 / 256, 256, 0, stream>>>(vec, mod_w, mod_b, modv);
  k_ln<<<L_SEQ, 256, 0, stream>>>(img, modv, xb);
  k_gemm<D_MODEL, NC1, 1><<<dim3(NC1 / 128, L_SEQ / 128), 256, 0, stream>>>(
      xb, lin1wb, lin1_b, qkvf, catb, nullptr, nullptr, nullptr);
  k_qkrope<<<(N_HEAD * L_SEQ) / 8, 256, 0, stream>>>(qkvf, pe, q_sc, k_sc, qb, kbv, vbv);
  k_attn<<<dim3(L_SEQ / 128, N_HEAD), 256, 0, stream>>>(qb, kbv, vbv, catb);
  k_gemm<KC2, D_MODEL, 2><<<dim3(D_MODEL / 128, L_SEQ / 128), 256, 0, stream>>>(
      catb, lin2wb, lin2_b, nullptr, nullptr, img, modv, (float*)d_out);
}
